// MultiHeadAttention_35373350650023
// MI455X (gfx1250) — matrix operand order repaired and checked by reading, not compiled, not run on hardware
//
#include <hip/hip_runtime.h>
#include <hip/hip_bf16.h>

// ---------------------------------------------------------------------------
// Problem constants (from reference)
// ---------------------------------------------------------------------------
#define D_EMBED   1536
#define N_HEADS   16
#define HD        96           // head dim
#define SEQ       2048
#define BATCH     2
#define NTOK      (BATCH * SEQ)          // 4096 rows for the projection GEMMs
#define ATT_SCALE 0.10206207261596577f   // 96^-0.5

typedef __bf16 bf16;
typedef __attribute__((ext_vector_type(4)))  __bf16 bf16x4;
typedef __attribute__((ext_vector_type(8)))  __bf16 bf16x8;
typedef __attribute__((ext_vector_type(16))) __bf16 v16bf;
typedef __attribute__((ext_vector_type(8)))  float  v8f;

__device__ __forceinline__ v8f zero8() {
  v8f z;
#pragma unroll
  for (int i = 0; i < 8; ++i) z[i] = 0.0f;
  return z;
}

// A-fragment (16x32 bf16, MxK): lane L: M = L&15, half = L>>4.
// elements 0..7  -> K = half*8 + e ; elements 8..15 -> K = 16 + half*8 + (e-8)
__device__ __forceinline__ v16bf load_frag_a(const bf16* __restrict__ base, int half) {
  bf16x8 lo = *(const bf16x8*)(base + half * 8);
  bf16x8 hi = *(const bf16x8*)(base + 16 + half * 8);
  v16bf a;
#pragma unroll
  for (int i = 0; i < 8; ++i) { a[i] = lo[i]; a[i + 8] = hi[i]; }
  return a;
}

// B-fragment (32x16 bf16, KxN): lane L: N = L&15, half = L>>4.
// elements 0..7  -> K = half*8 + e ; elements 8..15 -> K = 16 + half*8 + (e-8)
__device__ __forceinline__ v16bf load_frag_b(const bf16* __restrict__ base, int half) {
  bf16x8 lo = *(const bf16x8*)(base + half * 8);
  bf16x8 hi = *(const bf16x8*)(base + 16 + half * 8);
  v16bf b;
#pragma unroll
  for (int i = 0; i < 8; ++i) { b[i] = lo[i]; b[i + 8] = hi[i]; }
  return b;
}

#define WMMA_BF16(A, B, C) \
  __builtin_amdgcn_wmma_f32_16x16x32_bf16(false, (A), false, (B), (short)0, (C), false, false)

// ---------------------------------------------------------------------------
// fp32 -> bf16 conversion (vectorized x4)
// ---------------------------------------------------------------------------
__global__ __launch_bounds__(256) void cvt_f32_to_bf16(const float* __restrict__ src,
                                                       bf16* __restrict__ dst, int n) {
  int i = (blockIdx.x * blockDim.x + threadIdx.x) * 4;
  if (i + 3 < n) {
    float4 v = *(const float4*)(src + i);
    bf16x4 o;
    o[0] = (bf16)v.x; o[1] = (bf16)v.y; o[2] = (bf16)v.z; o[3] = (bf16)v.w;
    *(bf16x4*)(dst + i) = o;
  }
}

// ---------------------------------------------------------------------------
// GEMM: C[m,n] = (A[m,:] . W[n,:]) + bias[n], then *scale.
//   A : [NTOK x D] bf16 row-major, W : [D x D] bf16 row-major (acts as B = W^T)
// MODE 0: store bf16 to [B,H,S,hd]   (Q and K)
// MODE 1: store bf16 to [B,H,hd,S]   (V transposed per head)
// MODE 2: store fp32 to [NTOK x D]   (final output projection)
// Wave computes a 32x64 tile. K-loop is unrolled 4x so four independent
// load/WMMA batches are in flight (cross-iteration pipelining).
// ---------------------------------------------------------------------------
template <int MODE>
__global__ __launch_bounds__(128) void gemm_bf16(const bf16* __restrict__ A,
                                                 const bf16* __restrict__ W,
                                                 const float* __restrict__ bias,
                                                 void* __restrict__ out, float scale) {
  const int K = D_EMBED, N = D_EMBED;
  const int MT = NTOK / 32;                       // 128 row tiles (32 rows each)
  int wid  = blockIdx.x * 4 + (threadIdx.x >> 5);
  int lane = threadIdx.x & 31;
  int half = lane >> 4, ln = lane & 15;

  int mt = wid % MT;
  int nt = wid / MT;                              // 0..23 (64-wide col tiles)
  int n0 = nt * 64;

  v8f acc[2][4];
#pragma unroll
  for (int t = 0; t < 2; ++t)
#pragma unroll
    for (int f = 0; f < 4; ++f) acc[t][f] = zero8();

  const bf16* arow0 = A + (size_t)(mt * 32 + ln) * K;
  const bf16* arow1 = arow0 + (size_t)16 * K;
  const bf16* wbase = W + (size_t)(n0 + ln) * K;
#pragma unroll 4
  for (int k0 = 0; k0 < K; k0 += 32) {
    v16bf a0 = load_frag_a(arow0 + k0, half);
    v16bf a1 = load_frag_a(arow1 + k0, half);
#pragma unroll
    for (int f = 0; f < 4; ++f) {
      v16bf b = load_frag_b(wbase + (size_t)(f * 16) * K + k0, half);
      acc[0][f] = WMMA_BF16(a0, b, acc[0][f]);
      acc[1][f] = WMMA_BF16(a1, b, acc[1][f]);
    }
  }

#pragma unroll
  for (int t = 0; t < 2; ++t) {
#pragma unroll
    for (int f = 0; f < 4; ++f) {
      int n = n0 + f * 16 + ln;
      float bs = bias[n];
#pragma unroll
      for (int r = 0; r < 8; ++r) {
        float v = (acc[t][f][r] + bs) * scale;
        int m = mt * 32 + t * 16 + r + half * 8; // C layout: row = r + 8*half
        if (MODE == 2) {
          ((float*)out)[(size_t)m * N + n] = v;
        } else {
          int bb = m / SEQ, s = m % SEQ;
          int h = n / HD, d = n % HD;
          bf16 bv = (bf16)v;
          if (MODE == 0)
            ((bf16*)out)[(((size_t)(bb * N_HEADS + h) * SEQ) + s) * HD + d] = bv;
          else
            ((bf16*)out)[(((size_t)(bb * N_HEADS + h) * HD) + d) * SEQ + s] = bv;
        }
      }
    }
  }
}

// ---------------------------------------------------------------------------
// Online-softmax update for one 16x32 score tile pair; stages P into LDS.
//   sA: scores for keys t0..t0+15, sB: keys t0+16..t0+31 (C layout)
//   This lane's group (half) owns rows srow_base + half*8 + r.
// ---------------------------------------------------------------------------
__device__ __forceinline__ void softmax_update(v8f& sA, v8f& sB, int srow_base, int t0,
                                               int ln, int half,
                                               float* __restrict__ m_r,
                                               float* __restrict__ l_r,
                                               v8f* __restrict__ oacc,
                                               bf16* ldsrow) {
#pragma unroll
  for (int r = 0; r < 8; ++r) {
    int srow = srow_base + r + half * 8;
    float v0 = sA[r]; if (t0 + ln      > srow) v0 = -__builtin_inff();
    float v1 = sB[r]; if (t0 + 16 + ln > srow) v1 = -__builtin_inff();

    float mx = fmaxf(v0, v1);
#pragma unroll
    for (int off = 1; off < 16; off <<= 1) mx = fmaxf(mx, __shfl_xor(mx, off, 32));
    float mnew  = fmaxf(m_r[r], mx);
    float alpha = __expf(m_r[r] - mnew);

    float p0 = __expf(v0 - mnew);
    float p1 = __expf(v1 - mnew);
    float rs = p0 + p1;
#pragma unroll
    for (int off = 1; off < 16; off <<= 1) rs += __shfl_xor(rs, off, 32);

    l_r[r] = l_r[r] * alpha + rs;
    m_r[r] = mnew;

    ldsrow[(r + half * 8) * 32 + ln]      = (bf16)p0;
    ldsrow[(r + half * 8) * 32 + 16 + ln] = (bf16)p1;

#pragma unroll
    for (int f = 0; f < 6; ++f) oacc[f][r] *= alpha;
  }
}

// ---------------------------------------------------------------------------
// Flash attention (causal). One wave per (b,h, 32-query-row tile): two 16-row
// A-fragment sets share every K and V B-fragment.
//  - K fragments: rotated double-buffer (next pair issued before current WMMAs)
//  - V fragments: issued before the softmax, pinned with a sched_barrier so
//    global latency overlaps the exp/shuffle VALU chain.
//   Qb : [B,H,S,hd] bf16 (pre-scaled by ATT_SCALE)
//   Kb : [B,H,S,hd] bf16
//   Vt : [B,H,hd,S] bf16 (V transposed per head)
//   Ob : [B,S,H,hd] bf16 (A matrix for the output projection)
// ---------------------------------------------------------------------------
__global__ __launch_bounds__(128) void flash_attn(const bf16* __restrict__ Qb,
                                                  const bf16* __restrict__ Kb,
                                                  const bf16* __restrict__ Vt,
                                                  bf16* __restrict__ Ob) {
  __shared__ bf16 lds_p[4][2][16 * 32];          // per-wave, per-qtile P staging

  int wv   = threadIdx.x >> 5;
  int wid  = blockIdx.x * 4 + wv;
  int lane = threadIdx.x & 31;
  int half = lane >> 4, ln = lane & 15;

  const int QT = SEQ / 32;                       // 64 query tiles per head
  int qt = wid % QT;
  int bh = wid / QT;                             // 0..31
  int q0 = qt * 32;

  const bf16* qbase = Qb + (size_t)bh * SEQ * HD;
  const bf16* kbase = Kb + (size_t)bh * SEQ * HD;
  const bf16* vbase = Vt + (size_t)bh * HD * SEQ;

  // Q A-fragments for both 16-row tiles (row M = ln within each tile)
  v16bf aq[2][3];
  {
    const bf16* qrow0 = qbase + (size_t)(q0 + ln) * HD;
    const bf16* qrow1 = qbase + (size_t)(q0 + 16 + ln) * HD;
#pragma unroll
    for (int f = 0; f < 3; ++f) {
      aq[0][f] = load_frag_a(qrow0 + f * 32, half);
      aq[1][f] = load_frag_a(qrow1 + f * 32, half);
    }
  }

  float m_r[2][8], l_r[2][8];
#pragma unroll
  for (int t = 0; t < 2; ++t)
#pragma unroll
    for (int r = 0; r < 8; ++r) { m_r[t][r] = -__builtin_inff(); l_r[t][r] = 0.0f; }
  v8f o_acc[2][6];
#pragma unroll
  for (int t = 0; t < 2; ++t)
#pragma unroll
    for (int f = 0; f < 6; ++f) o_acc[t][f] = zero8();

  int nkb = q0 / 32 + 1;                         // causal: 32-key blocks
  for (int kb = 0; kb < nkb; ++kb) {
    int t0 = kb * 32;

    // Prefetch next key block's K/V rows (global_prefetch_b8)
    if (kb + 1 < nkb) {
      __builtin_prefetch(kbase + (size_t)(t0 + 32 + ln) * HD, 0, 1);
      __builtin_prefetch(vbase + (size_t)(ln * 6) * SEQ + t0 + 32, 0, 1);
    }

    const bf16* kp0 = kbase + (size_t)(t0 + ln) * HD;
    const bf16* kp1 = kbase + (size_t)(t0 + 16 + ln) * HD;

    // ---- scores: rotated double-buffer over 3 K-fragment pairs ----
    v8f s00 = zero8(), s01 = zero8(), s10 = zero8(), s11 = zero8();
    v16bf bk0 = load_frag_b(kp0, half);
    v16bf bk1 = load_frag_b(kp1, half);
#pragma unroll
    for (int f = 0; f < 3; ++f) {
      v16bf nk0, nk1;
      if (f < 2) {                                // issue next pair first
        nk0 = load_frag_b(kp0 + (f + 1) * 32, half);
        nk1 = load_frag_b(kp1 + (f + 1) * 32, half);
      }
      s00 = WMMA_BF16(aq[0][f], bk0, s00);
      s10 = WMMA_BF16(aq[1][f], bk0, s10);
      s01 = WMMA_BF16(aq[0][f], bk1, s01);
      s11 = WMMA_BF16(aq[1][f], bk1, s11);
      if (f < 2) { bk0 = nk0; bk1 = nk1; }
    }

    // ---- issue V B-fragment loads, pinned above the softmax ----
    v16bf bv[6];
#pragma unroll
    for (int f = 0; f < 6; ++f)
      bv[f] = load_frag_b(vbase + (size_t)(f * 16 + ln) * SEQ + t0, half);
    __builtin_amdgcn_sched_barrier(0);           // keep loads above softmax

    // ---- causal mask + online softmax per query tile ----
    softmax_update(s00, s01, q0,      t0, ln, half, m_r[0], l_r[0], o_acc[0], &lds_p[wv][0][0]);
    softmax_update(s10, s11, q0 + 16, t0, ln, half, m_r[1], l_r[1], o_acc[1], &lds_p[wv][1][0]);

    // wave-private LDS: enforce store->load ordering
    asm volatile("s_wait_dscnt 0" ::: "memory");

    // reload P tiles as A-fragments (row M = ln)
    v16bf ap0, ap1;
    {
      bf16x8 lo0 = *(bf16x8*)&lds_p[wv][0][ln * 32 + half * 8];
      bf16x8 hi0 = *(bf16x8*)&lds_p[wv][0][ln * 32 + 16 + half * 8];
      bf16x8 lo1 = *(bf16x8*)&lds_p[wv][1][ln * 32 + half * 8];
      bf16x8 hi1 = *(bf16x8*)&lds_p[wv][1][ln * 32 + 16 + half * 8];
#pragma unroll
      for (int i = 0; i < 8; ++i) {
        ap0[i] = lo0[i]; ap0[i + 8] = hi0[i];
        ap1[i] = lo1[i]; ap1[i + 8] = hi1[i];
      }
    }

    // ---- PV: V fragments shared by both query tiles (12 WMMAs) ----
#pragma unroll
    for (int f = 0; f < 6; ++f) {
      o_acc[0][f] = WMMA_BF16(ap0, bv[f], o_acc[0][f]);
      o_acc[1][f] = WMMA_BF16(ap1, bv[f], o_acc[1][f]);
    }
  }

  // ---- epilogue: normalize and store Ob[b,s,h,d] ----
  int bb = bh / N_HEADS, h = bh % N_HEADS;
#pragma unroll
  for (int t = 0; t < 2; ++t) {
#pragma unroll
    for (int r = 0; r < 8; ++r) {
      float inv = 1.0f / l_r[t][r];
      int s = q0 + t * 16 + r + half * 8;
      bf16* orow = Ob + ((size_t)(bb * SEQ + s) * N_HEADS + h) * HD;
#pragma unroll
      for (int f = 0; f < 6; ++f) orow[f * 16 + ln] = (bf16)(o_acc[t][f][r] * inv);
    }
  }
}

// ---------------------------------------------------------------------------
// Host-side launch
// ---------------------------------------------------------------------------
extern "C" void kernel_launch(void* const* d_in, const int* in_sizes, int n_in,
                              void* d_out, int out_size, void* d_ws, size_t ws_size,
                              hipStream_t stream) {
  const float* x  = (const float*)d_in[0];
  const float* Wq = (const float*)d_in[1];
  const float* bq = (const float*)d_in[2];
  const float* Wk = (const float*)d_in[3];
  const float* bk = (const float*)d_in[4];
  const float* Wv = (const float*)d_in[5];
  const float* bv = (const float*)d_in[6];
  const float* Wo = (const float*)d_in[7];
  const float* bo = (const float*)d_in[8];

  char* ws = (char*)d_ws;
  size_t off = 0;
  auto alloc = [&](size_t bytes) -> void* {
    void* p = ws + off;
    off += (bytes + 255) & ~(size_t)255;
    return p;
  };

  const size_t nX = (size_t)NTOK * D_EMBED;      // 6.29M elements
  const size_t nW = (size_t)D_EMBED * D_EMBED;   // 2.36M elements

  bf16* Xb  = (bf16*)alloc(nX * 2);
  bf16* Wqb = (bf16*)alloc(nW * 2);
  bf16* Wkb = (bf16*)alloc(nW * 2);
  bf16* Wvb = (bf16*)alloc(nW * 2);
  bf16* Wob = (bf16*)alloc(nW * 2);
  bf16* Qb  = (bf16*)alloc(nX * 2);              // [B,H,S,hd], pre-scaled
  bf16* Kb  = (bf16*)alloc(nX * 2);              // [B,H,S,hd]
  bf16* Vt  = (bf16*)alloc(nX * 2);              // [B,H,hd,S]
  bf16* Ob  = (bf16*)alloc(nX * 2);              // [B,S,H,hd]

  // 1) fp32 -> bf16 conversions
  cvt_f32_to_bf16<<<(int)(nX / 4 / 256), 256, 0, stream>>>(x, Xb, (int)nX);
  cvt_f32_to_bf16<<<(int)(nW / 4 / 256), 256, 0, stream>>>(Wq, Wqb, (int)nW);
  cvt_f32_to_bf16<<<(int)(nW / 4 / 256), 256, 0, stream>>>(Wk, Wkb, (int)nW);
  cvt_f32_to_bf16<<<(int)(nW / 4 / 256), 256, 0, stream>>>(Wv, Wvb, (int)nW);
  cvt_f32_to_bf16<<<(int)(nW / 4 / 256), 256, 0, stream>>>(Wo, Wob, (int)nW);

  // 2) QKV projections: (4096/32)*(1536/64) = 3072 waves -> 768 blocks
  const int gemm_blocks = (NTOK / 32) * (D_EMBED / 64) / 4;
  gemm_bf16<0><<<gemm_blocks, 128, 0, stream>>>(Xb, Wqb, bq, Qb, ATT_SCALE);
  gemm_bf16<0><<<gemm_blocks, 128, 0, stream>>>(Xb, Wkb, bk, Kb, 1.0f);
  gemm_bf16<1><<<gemm_blocks, 128, 0, stream>>>(Xb, Wvb, bv, Vt, 1.0f);

  // 3) causal flash attention: B*H*(S/32) = 2048 waves -> 512 blocks
  const int attn_blocks = BATCH * N_HEADS * (SEQ / 32) / 4;
  flash_attn<<<attn_blocks, 128, 0, stream>>>(Qb, Kb, Vt, Ob);

  // 4) output projection -> fp32 d_out
  gemm_bf16<2><<<gemm_blocks, 128, 0, stream>>>(Ob, Wob, bo, d_out, 1.0f);
}
